// rnn_single_crf_12695923327224
// MI455X (gfx1250) — matrix operand order repaired and checked by reading, not compiled, not run on hardware
//
#include <hip/hip_runtime.h>
#include <hip/hip_bf16.h>

typedef __attribute__((ext_vector_type(16))) __bf16 v16bf;
typedef __attribute__((ext_vector_type(8)))  float  v8f;

#define B_DIM 512
#define T_DIM 1024
#define NEGV (-10000.0f)
#define SA_STRIDE 72   // bf16 per row: 64 + 8 pad; 144B row stride keeps 16B alignment, shifts banks

#define LOG2E 1.4426950408889634f
#define LN2   0.6931471805599453f

// Raw single-instruction transcendentals (v_exp_f32 / v_log_f32 are base-2).
__device__ __forceinline__ float fast_exp(float x) {   // e^x
  return __builtin_amdgcn_exp2f(x * LOG2E);
}
__device__ __forceinline__ float fast_log(float x) {   // ln(x), x must be normal & finite
  return LN2 * __builtin_amdgcn_logf(x);
}
// Single v_med3_f32 clamp (avoids fmaxf's extra canonicalize op).
__device__ __forceinline__ float clamp_pos(float x) {
  return __builtin_amdgcn_fmed3f(x, 1.0e-30f, 3.0e38f);
}

// ---------------------------------------------------------------------------
// Kernel A: forward scan. One 16-batch-row slab per block; 2 waves per block,
// each wave owns 32 of the 64 destination tags (2 WMMA N-tiles).
//
// K-dimension of the prob-space GEMM is PERMUTED (j' = 32w + 2*l16 + tl) so
// that each lane's two exp values are adjacent -> one v_cvt_pk_bf16_f32 +
// one ds_store_b32 instead of two ds_store_b16. W's rows are permuted to
// match when sWt is built, so the contraction is unchanged.
// ---------------------------------------------------------------------------
__global__ __launch_bounds__(64) void crf_scan_kernel(
    const float* __restrict__ h, const float* __restrict__ trans,
    const int* __restrict__ y0, float* __restrict__ Zarr)
{
  __shared__ __align__(16) __bf16 sWt[64 * 64];        // exp(trans)^T (K-permuted), B-operand
  __shared__ __align__(16) __bf16 sA[16 * SA_STRIDE];  // per-step exp(score - rowmax) tile
  __shared__ float sMx[2][16];
  __shared__ float sMask[16];
  __shared__ float sFin[16 * 64];

  const int tid = threadIdx.x;
  const int w   = tid >> 5;        // wave id (0,1): owns N columns [w*32, w*32+32)
  const int l   = tid & 31;
  const int l16 = l & 15;
  const int hi  = l >> 4;          // lane half
  const int b0  = blockIdx.x * 16; // first batch row of this slab

  // sWt[n*64 + k'] = exp(trans[n][perm(k')]), perm(k') = (k'&32) | ((k'&1)<<4) | ((k'>>1)&15).
  // Column n's (permuted) K-run stays contiguous for the B-fragment b128 reads.
  for (int k = 0; k < 64; ++k) {
    int kk = (k & 32) | ((k & 1) << 4) | ((k >> 1) & 15);
    sWt[tid * 64 + k] = (__bf16)fast_exp(trans[tid * 64 + kk]);
  }
  __syncthreads();

  // B fragments (constant across scan). this code's contiguous K order (A uses the same):
  // lanes 0-15: K=0..15, lanes 16-31: K=16..31, lane%16 = column.
  v16bf Bfrag[2][2];
#pragma unroll
  for (int kb2 = 0; kb2 < 2; ++kb2) {
#pragma unroll
    for (int tl = 0; tl < 2; ++tl) {
      int n  = (2 * w + tl) * 16 + l16;
      int ks = kb2 * 32 + hi * 16;
      union { uint4 q[2]; v16bf v; } u;
      const uint4* p = (const uint4*)&sWt[n * 64 + ks];
      u.q[0] = p[0]; u.q[1] = p[1];
      Bfrag[kb2][tl] = u.v;
    }
  }

  // score in C/D layout: acc[tl][g] <-> (m = g + 8*hi, n = (2w+tl)*16 + l16)
  // All values stay FINITE (log input clamped below), so fma-blend masking is exact.
  v8f acc[2];
#pragma unroll
  for (int tl = 0; tl < 2; ++tl)
#pragma unroll
    for (int g = 0; g < 8; ++g) {
      int n = (2 * w + tl) * 16 + l16;
      acc[tl][g] = (n == 1) ? 0.0f : NEGV;  // SOS_IDX = 1
    }

  size_t baseRow[8];
#pragma unroll
  for (int g = 0; g < 8; ++g)
    baseRow[g] = ((size_t)(b0 + g + 8 * hi) * T_DIM) * 64 + (size_t)(w * 32 + l16);

  float hcur[16], hnext[16] = {};
#pragma unroll
  for (int tl = 0; tl < 2; ++tl)
#pragma unroll
    for (int g = 0; g < 8; ++g)
      hcur[tl * 8 + g] = h[baseRow[g] + (size_t)tl * 16];

  for (int t = 0; t < T_DIM; ++t) {
    // Prefetch h for t+1 (independent of the recurrence -> hides VMEM latency)
    if (t + 1 < T_DIM) {
#pragma unroll
      for (int tl = 0; tl < 2; ++tl)
#pragma unroll
        for (int g = 0; g < 8; ++g)
          hnext[tl * 8 + g] = h[baseRow[g] + (size_t)(t + 1) * 64 + (size_t)tl * 16];
    }
    if (w == 0 && l < 16)
      sMask[l] = (y0[(b0 + l) * (T_DIM + 1) + t + 1] > 0) ? 1.0f : 0.0f;

    // Row max over this wave's 32 columns (xor-shuffle within 16-lane group)
    float mxl[8];
#pragma unroll
    for (int g = 0; g < 8; ++g) mxl[g] = fmaxf(acc[0][g], acc[1][g]);
#pragma unroll
    for (int s = 1; s <= 8; s <<= 1)
#pragma unroll
      for (int g = 0; g < 8; ++g)
        mxl[g] = fmaxf(mxl[g], __shfl_xor(mxl[g], s, 32));
    if (l16 == 0)
#pragma unroll
      for (int g = 0; g < 8; ++g) sMx[w][g + 8 * hi] = mxl[g];
    __syncthreads();

    float mx[8], msk[8];
#pragma unroll
    for (int g = 0; g < 8; ++g) {
      int m  = g + 8 * hi;
      mx[g]  = fmaxf(sMx[0][m], sMx[1][m]);
      msk[g] = sMask[m];
    }

    // exp(score - rowmax) in [0,1] -> bf16 pair -> single b32 store per g.
    // Lane's two values (tl=0,1) land at permuted K = 32w + 2*l16 + {0,1}.
#pragma unroll
    for (int g = 0; g < 8; ++g) {
      int m = g + 8 * hi;
      union { __bf16 b[2]; unsigned int u; } pk;
      pk.b[0] = (__bf16)fast_exp(acc[0][g] - mx[g]);
      pk.b[1] = (__bf16)fast_exp(acc[1][g] - mx[g]);
      *(unsigned int*)&sA[m * SA_STRIDE + w * 32 + 2 * l16] = pk.u;
    }
    __syncthreads();

    // A fragments: 16x32 bf16 layout -> lane=row l16, contiguous K run kb+16*hi+0..15
    v16bf Afrag[2];
#pragma unroll
    for (int kb2 = 0; kb2 < 2; ++kb2) {
      union { uint4 q[2]; v16bf v; } u;
      int base = l16 * SA_STRIDE + kb2 * 32 + hi * 16;
      u.q[0] = *(const uint4*)&sA[base];
      u.q[1] = *(const uint4*)&sA[base + 8];
      Afrag[kb2] = u.v;
    }

#pragma unroll
    for (int tl = 0; tl < 2; ++tl) {
      v8f z = {};
      v8f c = __builtin_amdgcn_wmma_f32_16x16x32_bf16(false, Afrag[0], false, Bfrag[0][tl],
                                                      (short)0, z, false, false);
      c     = __builtin_amdgcn_wmma_f32_16x16x32_bf16(false, Afrag[1], false, Bfrag[1][tl],
                                                      (short)0, c, false, false);
#pragma unroll
      for (int g = 0; g < 8; ++g) {
        float u  = clamp_pos(c[g]);                // v_med3_f32: log stays finite & normal
        float sn = hcur[tl * 8 + g] + mx[g] + fast_log(u);
        // branch-free masked update: all operands finite, so blend is exact
        acc[tl][g] = fmaf(msk[g], sn - acc[tl][g], acc[tl][g]);
      }
    }
#pragma unroll
    for (int i = 0; i < 16; ++i) hcur[i] = hnext[i];
  }

  // Z[b] = logsumexp_n(score[b,n] + trans[EOS,n])
#pragma unroll
  for (int tl = 0; tl < 2; ++tl) {
    int n    = (2 * w + tl) * 16 + l16;
    float tr = trans[2 * 64 + n];  // EOS_IDX = 2
#pragma unroll
    for (int g = 0; g < 8; ++g) {
      int m = g + 8 * hi;
      sFin[m * 64 + n] = acc[tl][g] + tr;
    }
  }
  __syncthreads();
  if (w == 0 && l < 16) {
    float mxx = -3.0e38f;
    for (int j = 0; j < 64; ++j) mxx = fmaxf(mxx, sFin[l * 64 + j]);
    float s = 0.0f;
    for (int j = 0; j < 64; ++j) s += fast_exp(sFin[l * 64 + j] - mxx);
    Zarr[b0 + l] = mxx + fast_log(clamp_pos(s));  // s >= 1 anyway
  }
}

// ---------------------------------------------------------------------------
// Kernel B: gold path score. One wave per batch row; lanes stride over T.
// ---------------------------------------------------------------------------
__global__ __launch_bounds__(32) void crf_gold_kernel(
    const float* __restrict__ h, const float* __restrict__ trans,
    const int* __restrict__ y0, float* __restrict__ Gold)
{
  const int b = blockIdx.x, l = threadIdx.x;
  const int T1 = T_DIM + 1;
  float part = 0.0f;
  int   cnt  = 0;
  for (int t = l; t < T_DIM; t += 32) {
    int y1 = y0[b * T1 + t + 1];
    if (y1 > 0) {  // mask: y0[:,1:] > PAD
      int yp = y0[b * T1 + t];
      part += h[((size_t)b * T_DIM + t) * 64 + y1] + trans[y1 * 64 + yp];
      cnt++;
    }
  }
#pragma unroll
  for (int s = 16; s >= 1; s >>= 1) {
    part += __shfl_xor(part, s, 32);
    cnt  += __shfl_xor(cnt, s, 32);
  }
  if (l == 0) {
    int lastTag = y0[b * T1 + cnt];  // last_idx = sum(mask)
    Gold[b] = part + trans[2 * 64 + lastTag];
  }
}

// ---------------------------------------------------------------------------
// Kernel C: mean(Z - gold) -> scalar
// ---------------------------------------------------------------------------
__global__ __launch_bounds__(256) void crf_reduce_kernel(
    const float* __restrict__ Zarr, const float* __restrict__ Gold,
    float* __restrict__ out)
{
  __shared__ float red[256];
  int t = threadIdx.x;
  red[t] = (Zarr[t] - Gold[t]) + (Zarr[t + 256] - Gold[t + 256]);
  __syncthreads();
  for (int s = 128; s > 0; s >>= 1) {
    if (t < s) red[t] += red[t + s];
    __syncthreads();
  }
  if (t == 0) out[0] = red[0] * (1.0f / (float)B_DIM);
}

extern "C" void kernel_launch(void* const* d_in, const int* in_sizes, int n_in,
                              void* d_out, int out_size, void* d_ws, size_t ws_size,
                              hipStream_t stream) {
  (void)in_sizes; (void)n_in; (void)out_size; (void)ws_size;
  const float* h     = (const float*)d_in[0];
  const float* trans = (const float*)d_in[1];
  const int*   y0    = (const int*)d_in[2];
  float* out  = (float*)d_out;
  float* Zarr = (float*)d_ws;          // [512]
  float* Gold = Zarr + B_DIM;          // [512]

  crf_scan_kernel<<<dim3(B_DIM / 16), dim3(64), 0, stream>>>(h, trans, y0, Zarr);
  crf_gold_kernel<<<dim3(B_DIM), dim3(32), 0, stream>>>(h, trans, y0, Gold);
  crf_reduce_kernel<<<dim3(1), dim3(256), 0, stream>>>(Zarr, Gold, out);
}
